// Basic_RSNN_mem_88845693485072
// MI455X (gfx1250) — matrix operand order repaired and checked by reading, not compiled, not run on hardware
//
#include <hip/hip_runtime.h>
#include <hip/hip_bf16.h>

// ---------------------------------------------------------------------------
// RSNN forward for MI455X (gfx1250, wave32, WMMA 16x16x32 bf16).
// B=128, T=64, N_IN=1024, N_HID=4096, N_OUT=1024.
// Per step: inp = x_t@W1 + 0.1*spk@Wr ; LIF update ; spike ; o = 0.9*o + spk@Wo
// LDS-staged, double-buffered GEMM: block tile 128x64, per-wave 32x32.
// ---------------------------------------------------------------------------

typedef __bf16 bf16_t;
typedef __attribute__((ext_vector_type(16))) __bf16 v16bf;
typedef __attribute__((ext_vector_type(8)))  __bf16 v8bf;
typedef __attribute__((ext_vector_type(8)))  float  v8f;

#define B_SZ   128
#define T_SZ   64
#define NIN    1024
#define NHID   4096
#define NOUT   1024
#define TAU    0.9f
#define THRESH 0.5f
#define RECSC  0.1f

#define KC        64                 // K-chunk staged per LDS buffer
#define LDS_STR   72                 // padded row stride (elements): avoids the
                                     // 128B-stride ds_load_b128 bank conflicts
#define AS_CH     (128 * LDS_STR)    // A chunk: 128 rows x KC
#define BS_CH     (64  * LDS_STR)    // B chunk:  64 cols x KC

// ---- fragment loaders implementing the ISA §7.12.2 VGPR layouts -----------

// A-matrix 16x32 bf16 tile, row-major source with row stride lda (elements).
// lane<16: row=lane, K {0..7,16..23}; lanes>=16: K {8..15,24..31}.
__device__ __forceinline__ v16bf load_a_frag(const bf16_t* base, size_t lda,
                                             int lane) {
  int r  = lane & 15;
  int kb = (lane >> 4) << 3;                 // 0 or 8
  const bf16_t* p = base + (size_t)r * lda + kb;
  v8bf lo = *(const v8bf*)(p);
  v8bf hi = *(const v8bf*)(p + 16);
  v16bf a;
#pragma unroll
  for (int i = 0; i < 8; ++i) { a[i] = lo[i]; a[i + 8] = hi[i]; }
  return a;
}

// B-matrix 32x16 bf16 tile from W^T stored [N,K] row-major (stride ldk elems).
// lane<16: col=lane, K {0..7,16..23}; lanes>=16: col=lane-16, K {8..15,24..31}.
__device__ __forceinline__ v16bf load_b_frag(const bf16_t* base, size_t ldk,
                                             int lane) {
  int c  = lane & 15;
  int kb = (lane >> 4) << 3;                 // 0 or 8
  const bf16_t* p = base + (size_t)c * ldk + kb;
  v8bf lo = *(const v8bf*)(p);
  v8bf hi = *(const v8bf*)(p + 16);
  v16bf b;
#pragma unroll
  for (int i = 0; i < 8; ++i) { b[i] = lo[i]; b[i + 8] = hi[i]; }
  return b;
}

__device__ __forceinline__ v8f wmma_bf16(v16bf a, v16bf b, v8f c) {
  return __builtin_amdgcn_wmma_f32_16x16x32_bf16(false, a, false, b,
                                                 (short)0, c, false, false);
}

__device__ __forceinline__ v8f zero8() {
  v8f z;
#pragma unroll
  for (int i = 0; i < 8; ++i) z[i] = 0.0f;
  return z;
}

// ---- cooperative LDS staging (256 threads) --------------------------------

// A chunk: 128 rows x KC elems (16 KB): each thread copies 32 elems (64 B).
__device__ __forceinline__ void stage_a(bf16_t* dst, const bf16_t* __restrict__ src,
                                        size_t lda, int tid) {
  int row = tid >> 1;
  int ko  = (tid & 1) * 32;
  const v8bf* s = (const v8bf*)(src + (size_t)row * lda + ko);
  v8bf* d = (v8bf*)(dst + row * LDS_STR + ko);
#pragma unroll
  for (int i = 0; i < 4; ++i) d[i] = s[i];
}

// B chunk: 64 cols x KC elems (8 KB): each thread copies 16 elems (32 B).
__device__ __forceinline__ void stage_b(bf16_t* dst, const bf16_t* __restrict__ src,
                                        size_t ldb, int tid) {
  int row = tid >> 2;
  int ko  = (tid & 3) * 16;
  const v8bf* s = (const v8bf*)(src + (size_t)row * ldb + ko);
  v8bf* d = (v8bf*)(dst + row * LDS_STR + ko);
  d[0] = s[0];
  d[1] = s[1];
}

// ---- compute one staged chunk: load all 8 fragments, then 8 WMMAs ---------
// (straight-line issue keeps the accumulators pinned in place)

__device__ __forceinline__ void compute_chunk(const bf16_t* Ab, const bf16_t* Bb,
                                              int m0, int n0w, int lane,
                                              v8f& c00, v8f& c01, v8f& c10, v8f& c11) {
  v16bf a0k0 = load_a_frag(Ab + (m0 +  0) * LDS_STR +  0, LDS_STR, lane);
  v16bf a1k0 = load_a_frag(Ab + (m0 + 16) * LDS_STR +  0, LDS_STR, lane);
  v16bf b0k0 = load_b_frag(Bb + (n0w +  0) * LDS_STR +  0, LDS_STR, lane);
  v16bf b1k0 = load_b_frag(Bb + (n0w + 16) * LDS_STR +  0, LDS_STR, lane);
  v16bf a0k1 = load_a_frag(Ab + (m0 +  0) * LDS_STR + 32, LDS_STR, lane);
  v16bf a1k1 = load_a_frag(Ab + (m0 + 16) * LDS_STR + 32, LDS_STR, lane);
  v16bf b0k1 = load_b_frag(Bb + (n0w +  0) * LDS_STR + 32, LDS_STR, lane);
  v16bf b1k1 = load_b_frag(Bb + (n0w + 16) * LDS_STR + 32, LDS_STR, lane);

  c00 = wmma_bf16(a0k0, b0k0, c00);
  c01 = wmma_bf16(a0k0, b1k0, c01);
  c10 = wmma_bf16(a1k0, b0k0, c10);
  c11 = wmma_bf16(a1k0, b1k0, c11);
  c00 = wmma_bf16(a0k1, b0k1, c00);
  c01 = wmma_bf16(a0k1, b1k1, c01);
  c10 = wmma_bf16(a1k1, b0k1, c10);
  c11 = wmma_bf16(a1k1, b1k1, c11);
}

// ---- one GEMM phase: acc += Asrc[128,Ktot] @ Bsrc[64,Ktot]^T (this block's
// 64-col strip). Double-buffered; last chunk peeled so the steady-state body
// is unconditional (stage-next / compute-current / barrier). -----------------

__device__ __forceinline__ void gemm_phase(const bf16_t* __restrict__ Asrc, size_t lda,
                                           const bf16_t* __restrict__ Bsrc, size_t ldb,
                                           int Ktot, bf16_t* AsL, bf16_t* BsL,
                                           int tid, int lane, int m0, int n0w,
                                           v8f& c00, v8f& c01, v8f& c10, v8f& c11) {
  const int nch = Ktot / KC;
  stage_a(AsL, Asrc, lda, tid);
  stage_b(BsL, Bsrc, ldb, tid);
  __syncthreads();
  int buf = 0;
  for (int c = 0; c < nch - 1; ++c) {
    stage_a(AsL + (buf ^ 1) * AS_CH, Asrc + (size_t)(c + 1) * KC, lda, tid);
    stage_b(BsL + (buf ^ 1) * BS_CH, Bsrc + (size_t)(c + 1) * KC, ldb, tid);
    compute_chunk(AsL + buf * AS_CH, BsL + buf * BS_CH, m0, n0w, lane,
                  c00, c01, c10, c11);
    __syncthreads();
    buf ^= 1;
  }
  compute_chunk(AsL + buf * AS_CH, BsL + buf * BS_CH, m0, n0w, lane,
                c00, c01, c10, c11);
  __syncthreads();   // safe to re-stage AsL/BsL afterwards (next phase)
}

// ---- pre-pass kernels -----------------------------------------------------

__global__ void __launch_bounds__(256)
cvt_f32_to_bf16(const float* __restrict__ in, bf16_t* __restrict__ out, size_t n) {
  size_t i = (size_t)blockIdx.x * blockDim.x + threadIdx.x;
  size_t stride = (size_t)gridDim.x * blockDim.x;
  for (; i < n; i += stride) out[i] = (bf16_t)in[i];
}

// out[cols,rows](bf16) = scale * in[rows,cols](f32)^T, 32x32 LDS-tiled.
__global__ void __launch_bounds__(256)
transpose_cvt(const float* __restrict__ in, bf16_t* __restrict__ out,
              int rows, int cols, float scale) {
  __shared__ float tile[32][33];
  int tx = threadIdx.x & 31;
  int ty = threadIdx.x >> 5;                  // 0..7
  int c0 = blockIdx.x * 32;
  int r0 = blockIdx.y * 32;
#pragma unroll
  for (int i = 0; i < 32; i += 8)
    tile[ty + i][tx] = in[(size_t)(r0 + ty + i) * cols + c0 + tx];
  __syncthreads();
#pragma unroll
  for (int i = 0; i < 32; i += 8)
    out[(size_t)(c0 + ty + i) * rows + r0 + tx] = (bf16_t)(scale * tile[tx][ty + i]);
}

__global__ void __launch_bounds__(256)
zero_u32(unsigned int* __restrict__ p, size_t n) {
  size_t i = (size_t)blockIdx.x * blockDim.x + threadIdx.x;
  size_t stride = (size_t)gridDim.x * blockDim.x;
  for (; i < n; i += stride) p[i] = 0u;
}

// ---- epilogue helpers (C/D layout: VGPR r -> row r + 8*(lane>=16),
// col = lane&15 within the 16x16 tile) --------------------------------------

__device__ __forceinline__ void lif_tile(const v8f& acc, int row0, int col0,
                                         int lane,
                                         const bf16_t* __restrict__ Sold,
                                         bf16_t* __restrict__ Snew,
                                         float* __restrict__ hmem) {
  const int rr = (lane >> 4) * 8;
  const int cc = lane & 15;
#pragma unroll
  for (int r = 0; r < 8; ++r) {
    size_t idx = (size_t)(row0 + rr + r) * NHID + (col0 + cc);
    float mem = hmem[idx];
    float so  = (float)Sold[idx];                    // previous spike {0,1}
    float m2  = TAU * mem * (1.0f - so) + acc[r];    // decay + hard reset + input
    hmem[idx] = m2;
    Snew[idx] = (bf16_t)((m2 >= THRESH) ? 1.0f : 0.0f);
  }
}

__device__ __forceinline__ void out_tile(const v8f& acc, int row0, int col0,
                                         int lane, int t,
                                         float* __restrict__ omem,
                                         float* __restrict__ out) {
  const int rr = (lane >> 4) * 8;
  const int cc = lane & 15;
#pragma unroll
  for (int r = 0; r < 8; ++r) {
    int row = row0 + rr + r;
    int col = col0 + cc;
    size_t idx = (size_t)row * NOUT + col;
    float o = TAU * omem[idx] + acc[r];
    omem[idx] = o;
    out[((size_t)row * T_SZ + t) * NOUT + col] = o;
  }
}

// ---- per-timestep kernel A: inp = x_t@W1 + 0.1*S_old@Wr ; LIF ; spike -----
// Block: 256 threads = 8 waves (4 in M x 2 in N), block tile 128x64.
// Grid: NHID/64 = 64 blocks.

__global__ void __launch_bounds__(256)
rsnn_hidden_step(const bf16_t* __restrict__ xbf,   // [B, T, NIN] bf16
                 const bf16_t* __restrict__ W1T,   // [NHID, NIN] bf16
                 const bf16_t* __restrict__ WrT,   // [NHID, NHID] bf16 (0.1-folded)
                 const bf16_t* __restrict__ Sold,  // [B, NHID] bf16 spikes (prev)
                 bf16_t* __restrict__ Snew,        // [B, NHID] bf16 spikes (next)
                 float* __restrict__ hmem,         // [B, NHID] f32 membrane
                 int t) {
  __shared__ bf16_t AsL[2 * AS_CH];   // 36 KB
  __shared__ bf16_t BsL[2 * BS_CH];   // 18 KB

  const int tid  = threadIdx.x;
  const int lane = tid & 31;
  const int wid  = tid >> 5;                 // 0..7
  const int m0   = (wid & 3) * 32;           // wave row offset in block
  const int n0w  = (wid >> 2) * 32;          // wave col offset in block strip
  const int nblk = blockIdx.x * 64;          // block's column strip

  v8f c00 = zero8(), c01 = zero8(), c10 = zero8(), c11 = zero8();

  // G1: x_t @ W_fc1 (K = NIN). x row stride = T*NIN, column offset t*NIN.
  gemm_phase(xbf + (size_t)t * NIN, (size_t)T_SZ * NIN,
             W1T + (size_t)nblk * NIN, NIN,
             NIN, AsL, BsL, tid, lane, m0, n0w, c00, c01, c10, c11);

  // G2: spikes @ (0.1*W_rec) (K = NHID).
  gemm_phase(Sold, NHID,
             WrT + (size_t)nblk * NHID, NHID,
             NHID, AsL, BsL, tid, lane, m0, n0w, c00, c01, c10, c11);

  // Fused LIF epilogue over the wave's 32x32 tile.
  lif_tile(c00, m0 +  0, nblk + n0w +  0, lane, Sold, Snew, hmem);
  lif_tile(c01, m0 +  0, nblk + n0w + 16, lane, Sold, Snew, hmem);
  lif_tile(c10, m0 + 16, nblk + n0w +  0, lane, Sold, Snew, hmem);
  lif_tile(c11, m0 + 16, nblk + n0w + 16, lane, Sold, Snew, hmem);
}

// ---- per-timestep kernel B: o = 0.9*o + S@W_out ; out[:,t,:] = o -----------
// Same block structure; grid: NOUT/64 = 16 blocks.

__global__ void __launch_bounds__(256)
rsnn_output_step(const bf16_t* __restrict__ S,     // [B, NHID] bf16 spikes
                 const bf16_t* __restrict__ WoT,   // [NOUT, NHID] bf16
                 float* __restrict__ omem,         // [B, NOUT] f32
                 float* __restrict__ out,          // [B, T, NOUT] f32
                 int t) {
  __shared__ bf16_t AsL[2 * AS_CH];
  __shared__ bf16_t BsL[2 * BS_CH];

  const int tid  = threadIdx.x;
  const int lane = tid & 31;
  const int wid  = tid >> 5;
  const int m0   = (wid & 3) * 32;
  const int n0w  = (wid >> 2) * 32;
  const int nblk = blockIdx.x * 64;

  v8f c00 = zero8(), c01 = zero8(), c10 = zero8(), c11 = zero8();

  gemm_phase(S, NHID,
             WoT + (size_t)nblk * NHID, NHID,
             NHID, AsL, BsL, tid, lane, m0, n0w, c00, c01, c10, c11);

  out_tile(c00, m0 +  0, nblk + n0w +  0, lane, t, omem, out);
  out_tile(c01, m0 +  0, nblk + n0w + 16, lane, t, omem, out);
  out_tile(c10, m0 + 16, nblk + n0w +  0, lane, t, omem, out);
  out_tile(c11, m0 + 16, nblk + n0w + 16, lane, t, omem, out);
}

// ---------------------------------------------------------------------------

extern "C" void kernel_launch(void* const* d_in, const int* in_sizes, int n_in,
                              void* d_out, int out_size, void* d_ws, size_t ws_size,
                              hipStream_t stream) {
  const float* x  = (const float*)d_in[0];   // [B, T, NIN]
  const float* W1 = (const float*)d_in[1];   // [NIN, NHID]
  const float* Wr = (const float*)d_in[2];   // [NHID, NHID] (zero diagonal)
  const float* Wo = (const float*)d_in[3];   // [NHID, NOUT]
  float* out = (float*)d_out;                // [B, T, NOUT]

  char* ws = (char*)d_ws;
  size_t off = 0;
  auto wsalloc = [&](size_t bytes) -> void* {
    void* p = ws + off;
    off += (bytes + 255) & ~(size_t)255;
    return p;
  };

  bf16_t* xbf = (bf16_t*)wsalloc((size_t)B_SZ * T_SZ * NIN * 2);  // 16 MB
  bf16_t* W1T = (bf16_t*)wsalloc((size_t)NIN * NHID * 2);         //  8 MB
  bf16_t* WrT = (bf16_t*)wsalloc((size_t)NHID * NHID * 2);        // 32 MB
  bf16_t* WoT = (bf16_t*)wsalloc((size_t)NHID * NOUT * 2);        //  8 MB
  // keep the state buffers contiguous so one zero-fill covers them all
  bf16_t* S0   = (bf16_t*)wsalloc((size_t)B_SZ * NHID * 2);       //  1 MB
  bf16_t* S1   = (bf16_t*)wsalloc((size_t)B_SZ * NHID * 2);       //  1 MB
  float*  hmem = (float*)wsalloc((size_t)B_SZ * NHID * 4);        //  2 MB
  float*  omem = (float*)wsalloc((size_t)B_SZ * NOUT * 4);        // .5 MB

  // --- pre-pass: bf16 conversions / transposes (weights become [N,K]) ------
  cvt_f32_to_bf16<<<2048, 256, 0, stream>>>(x, xbf, (size_t)B_SZ * T_SZ * NIN);
  transpose_cvt<<<dim3(NHID / 32, NIN / 32), 256, 0, stream>>>(W1, W1T, NIN, NHID, 1.0f);
  transpose_cvt<<<dim3(NHID / 32, NHID / 32), 256, 0, stream>>>(Wr, WrT, NHID, NHID, RECSC);
  transpose_cvt<<<dim3(NOUT / 32, NHID / 32), 256, 0, stream>>>(Wo, WoT, NHID, NOUT, 1.0f);

  size_t zero_words = ((size_t)B_SZ * NHID * 2 * 2 +   // S0 + S1
                       (size_t)B_SZ * NHID * 4 +       // hmem
                       (size_t)B_SZ * NOUT * 4) / 4;   // omem
  zero_u32<<<1024, 256, 0, stream>>>((unsigned int*)S0, zero_words);

  // --- sequential timestep loop (ping-pong spike buffers) ------------------
  for (int t = 0; t < T_SZ; ++t) {
    bf16_t* Sold = (t & 1) ? S1 : S0;
    bf16_t* Snew = (t & 1) ? S0 : S1;
    rsnn_hidden_step<<<NHID / 64, 256, 0, stream>>>(xbf, W1T, WrT, Sold, Snew, hmem, t);
    rsnn_output_step<<<NOUT / 64, 256, 0, stream>>>(Snew, WoT, omem, out, t);
  }
}
